// DynamicConvolution_69140383531364
// MI455X (gfx1250) — matrix operand order repaired and checked by reading, not compiled, not run on hardware
//
#include <hip/hip_runtime.h>

typedef __attribute__((ext_vector_type(16))) __bf16 v16bf;
typedef __attribute__((ext_vector_type(8)))  float  v8f;

#define HW    512
#define NPIX  (HW * HW)          // 262144
#define CIN   32
#define COUT  32
#define NB    16
#define NKER  4
#define KD    288                // CIN * 9, K ordering: (ky*3+kx)*32 + c

// workspace layout (bytes)
#define OFF_POOLED 0                                  // 512 floats
#define OFF_DYNA   4096                               // NB*COUT*KD bf16 = 294912 B
#define OFF_STATS  (4096 + NB * COUT * KD * 2)        // 64 floats (sum, sumsq)
#define OFF_SCALE  (OFF_STATS + 256)                  // 64 floats (scale, shift)

// ---------------------------------------------------------------------------
// 1) Global average pool of context_features -> pooled[B*CIN]
// ---------------------------------------------------------------------------
__global__ void pool_kernel(const float* __restrict__ ctx, float* __restrict__ pooled) {
    __shared__ float red[256];
    const int p   = blockIdx.x;                       // (b*CIN + c), 512 blocks
    const int tid = threadIdx.x;
    const float4* src = (const float4*)(ctx + (size_t)p * NPIX);
    float s = 0.f;
    #pragma unroll 4
    for (int j = 0; j < NPIX / 4 / 256; ++j) {
        float4 v = src[j * 256 + tid];
        s += v.x + v.y + v.z + v.w;
    }
    red[tid] = s;
    __syncthreads();
    for (int off = 128; off > 0; off >>= 1) {
        if (tid < off) red[tid] += red[tid + off];
        __syncthreads();
    }
    if (tid == 0) pooled[p] = red[0] * (1.0f / (float)NPIX);
}

// ---------------------------------------------------------------------------
// 2) Attention softmax + per-sample dynamic kernel, emitted as bf16 GEMM-A
//    A[b][o][ (ky*3+kx)*32 + c ].  Also zeroes the BN stats accumulators.
// ---------------------------------------------------------------------------
__global__ void dynk_kernel(const float* __restrict__ pooled,
                            const float* __restrict__ kernels,
                            const float* __restrict__ attn_w,
                            const float* __restrict__ attn_b,
                            __bf16* __restrict__ dynA,
                            float* __restrict__ stats) {
    const int b = blockIdx.x, tid = threadIdx.x;
    if (b == 0 && tid < 64) stats[tid] = 0.f;         // deterministic per-call zero
    __shared__ float lg[NKER];
    __shared__ float w[NKER];
    if (tid < NKER) {
        float l = attn_b[tid];
        #pragma unroll
        for (int c = 0; c < CIN; ++c) l += pooled[b * CIN + c] * attn_w[tid * CIN + c];
        lg[tid] = l;
    }
    __syncthreads();
    if (tid == 0) {
        float m = lg[0];
        #pragma unroll
        for (int n = 1; n < NKER; ++n) m = fmaxf(m, lg[n]);
        float e[NKER], s = 0.f;
        #pragma unroll
        for (int n = 0; n < NKER; ++n) { e[n] = __expf(lg[n] - m); s += e[n]; }
        #pragma unroll
        for (int n = 0; n < NKER; ++n) w[n] = e[n] / s;
    }
    __syncthreads();
    for (int idx = tid; idx < COUT * KD; idx += blockDim.x) {
        const int o  = idx / KD;
        const int k  = idx - o * KD;
        const int kk = k >> 5;                        // filter tap (ky*3+kx)
        const int c  = k & 31;                        // input channel
        float acc = 0.f;
        #pragma unroll
        for (int n = 0; n < NKER; ++n)
            acc += w[n] * kernels[(((n * COUT + o) * CIN + c) * 9) + kk];
        dynA[(size_t)b * COUT * KD + idx] = (__bf16)acc;
    }
}

// ---------------------------------------------------------------------------
// 3) Implicit-GEMM conv via v_wmma_f32_16x16x32_bf16.
//    Block = (x-tile of 128 px, row y, batch b); 8 waves, each wave owns a
//    16-wide pixel slice and both 16-row M tiles (C_OUT = 32).
//    Filter tile staged via GLOBAL_LOAD_ASYNC_TO_LDS_B128 (ASYNCcnt path),
//    overlapped with the f32->bf16 halo staging on LOADcnt/DScnt.
// ---------------------------------------------------------------------------
__global__ __launch_bounds__(256)
void conv_wmma_kernel(const float* __restrict__ x,
                      const __bf16* __restrict__ dynA,
                      float* __restrict__ out) {
    // xs layout: [row 0..2][col 0..129][c 0..31] bf16 -> channel contiguous
    __shared__ __align__(16) __bf16 xs[3 * 130 * CIN];   // 24960 B
    __shared__ __align__(16) __bf16 As[COUT * KD];       // 18432 B

    const int b  = blockIdx.z;
    const int y  = blockIdx.y;
    const int x0 = blockIdx.x * 128;
    const int tid = threadIdx.x;

    // ---- stage dynamic filter for this batch: async DMA global -> LDS ----
    {
        const unsigned long long gbase =
            (unsigned long long)(uintptr_t)(dynA + (size_t)b * COUT * KD);
        // low 32 bits of the generic pointer = LDS byte offset of As[]
        const unsigned lds_base = (unsigned)(uintptr_t)(&As[0]);
        #pragma unroll
        for (int j = 0; j < 5; ++j) {                 // 1152 x 16B chunks
            const int idx = tid + j * 256;
            if (idx < (COUT * KD * 2) / 16) {
                const unsigned off = (unsigned)idx * 16u;
                const unsigned ldsa = lds_base + off;
                asm volatile("global_load_async_to_lds_b128 %0, %1, %2"
                             :
                             : "v"(ldsa), "v"(off), "s"(gbase)
                             : "memory");
            }
        }
    }

    // ---- stage x halo tile (rows y-1..y+1, cols x0-1..x0+128) as bf16 ----
    const size_t xb = (size_t)b * CIN * NPIX;
    for (int i = tid; i < CIN * 3 * 130; i += 256) {
        const int c   = i / 390;                      // coalesced along columns
        const int t   = i - c * 390;                  // r*130 + col
        const int r   = t / 130;
        const int col = t - r * 130;
        const int gy  = y + r - 1;
        const int gx  = x0 + col - 1;
        float v = 0.f;
        if (gy >= 0 && gy < HW && gx >= 0 && gx < HW)
            v = x[xb + (size_t)c * NPIX + (size_t)gy * HW + gx];
        xs[t * CIN + c] = (__bf16)v;
    }

    asm volatile("s_wait_asynccnt 0x0" ::: "memory");  // filter DMA complete
    __syncthreads();

    const int wave  = tid >> 5;
    const int lane  = tid & 31;
    const int g     = lane >> 4;                      // half-wave group
    const int n     = lane & 15;                      // N column within tile
    const int wbase = wave * 16;                      // wave's pixel slice

    v8f c0 = {};                                      // M rows 0..15
    v8f c1 = {};                                      // M rows 16..31

    #pragma unroll
    for (int kk = 0; kk < 9; ++kk) {                  // filter taps = K chunks of 32
        const int ky = kk / 3, kx = kk % 3;

        // B fragment (32x16, K=c contiguous in LDS): element e -> K = g*16 + e
        v16bf bfrag;
        const __bf16* bsrc = &xs[((ky * 130) + (wbase + n + kx)) * CIN + g * 16];
        #pragma unroll
        for (int e = 0; e < 16; ++e) bfrag[e] = bsrc[e];

        // A fragments (16x32, contiguous K order, not ISA table order):
        // element e: K = g*16 + e (matches B)
        v16bf a0, a1;
        #pragma unroll
        for (int e = 0; e < 16; ++e) {
            const int v    = e >> 1;
            const int kidx = (g << 4) + e;
            a0[e] = As[n        * KD + kk * 32 + kidx];
            a1[e] = As[(n + 16) * KD + kk * 32 + kidx];
        }

        c0 = __builtin_amdgcn_wmma_f32_16x16x32_bf16(false, a0, false, bfrag,
                                                     (short)0, c0, false, false);
        c1 = __builtin_amdgcn_wmma_f32_16x16x32_bf16(false, a1, false, bfrag,
                                                     (short)0, c1, false, false);
    }

    // D layout: VGPR r, half-wave g -> M = g*8 + r, N = lane%16
    const size_t ob = (size_t)b * COUT * NPIX + (size_t)y * HW + (x0 + wbase + n);
    #pragma unroll
    for (int r = 0; r < 8; ++r) {
        const int o = g * 8 + r;
        out[ob + (size_t)o        * NPIX] = c0[r];
        out[ob + (size_t)(o + 16) * NPIX] = c1[r];
    }
}

// ---------------------------------------------------------------------------
// 4) BN batch statistics: per-channel sum / sumsq over (B,H,W)
// ---------------------------------------------------------------------------
__global__ void bnstat_kernel(const float* __restrict__ out, float* __restrict__ stats) {
    __shared__ float rs[256], rq[256];
    const int bi   = blockIdx.x;                      // 2048 = 32 ch * 64 parts
    const int ch   = bi >> 6;
    const int part = bi & 63;
    const int b    = part >> 2;
    const size_t base = ((size_t)b * COUT + ch) * NPIX + (size_t)(part & 3) * 65536;
    const float4* src = (const float4*)(out + base);
    const int tid = threadIdx.x;
    float s = 0.f, q = 0.f;
    for (int j = 0; j < 64; ++j) {
        if (j + 8 < 64) __builtin_prefetch(&src[(j + 8) * 256 + tid], 0, 0);
        float4 v = src[j * 256 + tid];
        s += v.x + v.y + v.z + v.w;
        q += v.x * v.x + v.y * v.y + v.z * v.z + v.w * v.w;
    }
    rs[tid] = s; rq[tid] = q;
    __syncthreads();
    for (int off = 128; off > 0; off >>= 1) {
        if (tid < off) { rs[tid] += rs[tid + off]; rq[tid] += rq[tid + off]; }
        __syncthreads();
    }
    if (tid == 0) {
        atomicAdd(&stats[ch], rs[0]);
        atomicAdd(&stats[COUT + ch], rq[0]);
    }
}

// ---------------------------------------------------------------------------
// 5) Finalize BN: scale = gamma*rsqrt(var+eps), shift = beta - mean*scale
// ---------------------------------------------------------------------------
__global__ void bnfinal_kernel(const float* __restrict__ stats,
                               const float* __restrict__ gamma,
                               const float* __restrict__ beta,
                               float* __restrict__ ss) {
    const int c = threadIdx.x;
    if (c < COUT) {
        const float invN = 1.0f / ((float)NB * (float)NPIX);
        const float mean = stats[c] * invN;
        const float var  = stats[COUT + c] * invN - mean * mean;
        const float inv  = rsqrtf(var + 1e-5f);
        const float sc   = inv * gamma[c];
        ss[c]        = sc;
        ss[COUT + c] = beta[c] - mean * sc;
    }
}

// ---------------------------------------------------------------------------
// 6) Apply BN in place on the conv output
// ---------------------------------------------------------------------------
__global__ void bnapply_kernel(float* __restrict__ out, const float* __restrict__ ss) {
    const size_t i4 = (size_t)blockIdx.x * 256 + threadIdx.x;  // float4 index
    const int ch = (int)(i4 >> 16) & 31;                        // (i4*4 / NPIX) % 32
    float4* p = (float4*)out + i4;
    float4 v = *p;
    const float sc = ss[ch], sh = ss[COUT + ch];
    v.x = v.x * sc + sh;
    v.y = v.y * sc + sh;
    v.z = v.z * sc + sh;
    v.w = v.w * sc + sh;
    *p = v;
}

// ---------------------------------------------------------------------------
extern "C" void kernel_launch(void* const* d_in, const int* in_sizes, int n_in,
                              void* d_out, int out_size, void* d_ws, size_t ws_size,
                              hipStream_t stream) {
    const float* x       = (const float*)d_in[0];
    const float* ctx     = (const float*)d_in[1];
    const float* kernels = (const float*)d_in[2];
    const float* attn_w  = (const float*)d_in[3];
    const float* attn_b  = (const float*)d_in[4];
    const float* gamma   = (const float*)d_in[5];
    const float* beta    = (const float*)d_in[6];
    float* out = (float*)d_out;

    char* ws = (char*)d_ws;
    float*  pooled = (float*)(ws + OFF_POOLED);
    __bf16* dynA   = (__bf16*)(ws + OFF_DYNA);
    float*  stats  = (float*)(ws + OFF_STATS);
    float*  ss     = (float*)(ws + OFF_SCALE);

    pool_kernel<<<NB * CIN, 256, 0, stream>>>(ctx, pooled);
    dynk_kernel<<<NB, 256, 0, stream>>>(pooled, kernels, attn_w, attn_b, dynA, stats);

    dim3 grid(HW / 128, HW, NB);                      // (4, 512, 16)
    conv_wmma_kernel<<<grid, 256, 0, stream>>>(x, dynA, out);

    bnstat_kernel<<<COUT * 64, 256, 0, stream>>>(out, stats);
    bnfinal_kernel<<<1, 32, 0, stream>>>(stats, gamma, beta, ss);
    bnapply_kernel<<<(NB * COUT * NPIX) / 4 / 256, 256, 0, stream>>>(out, ss);
}
